// GroupedQueryAttention_68186900791937
// MI455X (gfx1250) — matrix operand order repaired and checked by reading, not compiled, not run on hardware
//
#include <hip/hip_runtime.h>
#include <cstdint>

// ---------------------------------------------------------------------------
// GQA forward for MI455X (gfx1250, wave32, WMMA, async global->LDS).
// fp32 -> bf16 (weights pre-transposed), QKV GEMM (wmma bf16, async-LDS
// double buffered), RoPE scatter (V stored hd-major), flash-attention
// (wmma bf16, online softmax, async-LDS KV tiles), out GEMM (wmma -> fp32).
// ---------------------------------------------------------------------------

typedef __bf16 bf16;
typedef __attribute__((ext_vector_type(16))) __bf16 v16bf;
typedef __attribute__((ext_vector_type(8)))  __bf16 v8bf;
typedef __attribute__((ext_vector_type(8)))  float  v8f;
typedef int v4i_ __attribute__((vector_size(16)));   // matches async builtin pointee

constexpr int B_ = 2, S_ = 2048, D_ = 4096, H_ = 32, G_ = 8, HD_ = 128;
constexpr int GROUP_ = H_ / G_;                    // 4
constexpr int NQKV_  = H_ * HD_ + 2 * G_ * HD_;    // 6144
constexpr float SCALE_ = 0.08838834764831845f;     // 128^-0.5

#if defined(__AMDGCN__) && __has_builtin(__builtin_amdgcn_global_load_async_to_lds_b128) && \
    __has_builtin(__builtin_amdgcn_s_wait_asynccnt)
#define HAS_ASYNC_LDS 1
#else
#define HAS_ASYNC_LDS 0
#endif

__device__ __forceinline__ bf16 f2bf(float f) {
  uint32_t u = __builtin_bit_cast(uint32_t, f);
  u += 0x7fffu + ((u >> 16) & 1u);                 // round-to-nearest-even
  uint16_t h = (uint16_t)(u >> 16);
  return __builtin_bit_cast(bf16, h);
}
__device__ __forceinline__ float bf2f(bf16 b) {
  uint32_t u = ((uint32_t)__builtin_bit_cast(uint16_t, b)) << 16;
  return __builtin_bit_cast(float, u);
}

// 16B global -> LDS copy. Async path uses GLOBAL_LOAD_ASYNC_TO_LDS_B128
// (ASYNCcnt-tracked, no VGPR staging); fallback is a sync uint4 round-trip.
__device__ __forceinline__ void async_copy16(const bf16* gsrc, bf16* ldst) {
#if HAS_ASYNC_LDS
  __builtin_amdgcn_global_load_async_to_lds_b128(
      (__attribute__((address_space(1))) v4i_*)(v4i_*)const_cast<bf16*>(gsrc),
      (__attribute__((address_space(3))) v4i_*)(v4i_*)ldst, 0, 0);
#else
  *(uint4*)ldst = *(const uint4*)gsrc;
#endif
}
__device__ __forceinline__ void wait_async_all() {
#if HAS_ASYNC_LDS
  __builtin_amdgcn_s_wait_asynccnt(0);
#endif
}

// A-matrix 16x32 bf16 fragment (ISA 7.12.2): lanes 0-15 hold K={0..7,16..23},
// lanes 16-31 hold K={8..15,24..31}, row M = lane&15.
__device__ __forceinline__ v16bf load_frag_a(const bf16* row, int half) {
  union { v16bf v; v8bf h[2]; } u;
  u.h[0] = *(const v8bf*)(row + half * 8);
  u.h[1] = *(const v8bf*)(row + 16 + half * 8);
  return u.v;
}
// B-matrix 32x16 bf16 fragment: lanes 0-15 hold K={0..7,16..23}, lanes 16-31
// hold K={8..15,24..31}, column N = lane&15 (data stored N-major).
__device__ __forceinline__ v16bf load_frag_b(const bf16* row, int half) {
  union { v16bf v; v8bf h[2]; } u;
  u.h[0] = *(const v8bf*)(row + half * 8);
  u.h[1] = *(const v8bf*)(row + 16 + half * 8);
  return u.v;
}

__device__ __forceinline__ float half_max(float v) {
  #pragma unroll
  for (int m = 8; m; m >>= 1) v = fmaxf(v, __shfl_xor(v, m, 32));
  return v;   // reduced over the 16-lane half (bit4 untouched)
}
__device__ __forceinline__ float half_sum(float v) {
  #pragma unroll
  for (int m = 8; m; m >>= 1) v += __shfl_xor(v, m, 32);
  return v;
}

// --------------------------- fp32 -> bf16 convert ---------------------------
__global__ void cvt_kernel(const float* __restrict__ src, bf16* __restrict__ dst,
                           int n) {
  int i = blockIdx.x * blockDim.x + threadIdx.x;
  if (i < n) dst[i] = f2bf(src[i]);
}
// convert + transpose: dst[(rowoff + n) * ldk + k] = src[k * N + n]
__global__ void cvtT_kernel(const float* __restrict__ src, bf16* __restrict__ dst,
                            int K, int N, int ldk, int rowoff) {
  long long i = (long long)blockIdx.x * blockDim.x + threadIdx.x;
  if (i >= (long long)K * N) return;
  int k = (int)(i / N), n = (int)(i % N);
  dst[(long long)(rowoff + n) * ldk + k] = f2bf(src[i]);
}

// ------------------------------- WMMA GEMM ----------------------------------
// C[M,N] = A[M,K] * Bt[N,K]^T, bf16 in, f32 accumulate. 128x128x32 block tile,
// 256 threads = 8 waves (2x4); each wave: 64x32 tile = 4x2 WMMA tiles.
// Both LDS tiles are filled with async b128 copies, double buffered.
__global__ __launch_bounds__(256)
void gemm_bf16_kernel(const bf16* __restrict__ A, const bf16* __restrict__ Bt,
                      void* __restrict__ C, int M, int N, int K, int c_is_f32) {
  constexpr int BK  = 32;
  constexpr int LDA = 40;                          // padded stride (16B-aligned)
  __shared__ alignas(16) bf16 As[2][128 * LDA];    // row-major [m][k]
  __shared__ alignas(16) bf16 Bs[2][128 * LDA];    // N-major  [n][k]

  const int tid  = threadIdx.x;
  const int lane = tid & 31;
  const int half = (lane >> 4) & 1;
  const int lm   = lane & 15;
  const int wid  = tid >> 5;
  const int wm   = wid >> 2;                       // 0..1
  const int wn   = wid & 3;                        // 0..3
  const int m0   = blockIdx.y * 128;
  const int n0   = blockIdx.x * 128;

  const v8f vzero = {};
  v8f acc[4][2];
  #pragma unroll
  for (int i = 0; i < 4; ++i)
    #pragma unroll
    for (int j = 0; j < 2; ++j) acc[i][j] = vzero;

  // 128x32 bf16 tile = 512 x 16B chunks; 2 chunks per thread per tile.
  auto issue_tile = [&](int kt, int buf) {
    const int k0 = kt * BK;
    #pragma unroll
    for (int u = 0; u < 2; ++u) {
      const int c   = u * 256 + tid;
      const int row = c >> 2, off = (c & 3) * 8;
      async_copy16(A  + (m0 + row) * K + k0 + off, &As[buf][row * LDA + off]);
      async_copy16(Bt + (n0 + row) * K + k0 + off, &Bs[buf][row * LDA + off]);
    }
  };

  const int nkt = K / BK;
  issue_tile(0, 0);
  for (int kt = 0; kt < nkt; ++kt) {
    const int cb = kt & 1;
    wait_async_all();
    __syncthreads();                         // tile kt visible to all waves
    if (kt + 1 < nkt) issue_tile(kt + 1, cb ^ 1);  // overlap with compute

    v16bf af[4], bfr[2];
    #pragma unroll
    for (int i = 0; i < 4; ++i)
      af[i] = load_frag_a(&As[cb][(wm * 64 + i * 16 + lm) * LDA], half);
    #pragma unroll
    for (int j = 0; j < 2; ++j)
      bfr[j] = load_frag_b(&Bs[cb][(wn * 32 + j * 16 + lm) * LDA], half);
    #pragma unroll
    for (int i = 0; i < 4; ++i)
      #pragma unroll
      for (int j = 0; j < 2; ++j)
        acc[i][j] = __builtin_amdgcn_wmma_f32_16x16x32_bf16(
            false, af[i], false, bfr[j], (short)0, acc[i][j], false, false);
  }

  // C layout: VGPR r -> m = r + 8*half, n = lane&15
  #pragma unroll
  for (int i = 0; i < 4; ++i) {
    #pragma unroll
    for (int j = 0; j < 2; ++j) {
      const int n = n0 + wn * 32 + j * 16 + lm;
      #pragma unroll
      for (int r = 0; r < 8; ++r) {
        const int m = m0 + wm * 64 + i * 16 + r + 8 * half;
        if (c_is_f32) ((float*)C)[m * N + n] = acc[i][j][r];
        else          ((bf16*)C)[m * N + n]  = f2bf(acc[i][j][r]);
      }
    }
  }
}

// ----------------------- RoPE + scale + head scatter ------------------------
// Q: [b][h][s][hd] (rope, *SCALE);  K: [b][g][s][hd] (rope);
// V: [b][g][hd][S] (hd-major so attention can async-copy V^T tiles directly).
__global__ void rope_scatter_kernel(const bf16* __restrict__ qkv,
                                    const float* __restrict__ cosb,
                                    const float* __restrict__ sinb,
                                    bf16* __restrict__ Qh, bf16* __restrict__ Kh,
                                    bf16* __restrict__ Vh) {
  long long i = (long long)blockIdx.x * blockDim.x + threadIdx.x;
  if (i >= (long long)B_ * S_ * NQKV_) return;
  const int c = (int)(i % NQKV_);
  const int t = (int)(i / NQKV_);
  const int s = t % S_;
  const int b = t / S_;
  const bf16* row = qkv + (long long)t * NQKV_;
  if (c < H_ * HD_) {
    const int h = c >> 7, d = c & 127;
    const float v  = bf2f(row[c]);
    const float o  = bf2f(row[(d < 64) ? (c + 64) : (c - 64)]);
    const float cs = cosb[s * HD_ + d], sn = sinb[s * HD_ + d];
    const float rv = (d < 64) ? (v * cs - o * sn) : (v * cs + o * sn);
    Qh[((b * H_ + h) * S_ + s) * HD_ + d] = f2bf(rv * SCALE_);
  } else if (c < H_ * HD_ + G_ * HD_) {
    const int cc = c - H_ * HD_;
    const int g = cc >> 7, d = cc & 127;
    const float v  = bf2f(row[c]);
    const float o  = bf2f(row[(d < 64) ? (c + 64) : (c - 64)]);
    const float cs = cosb[s * HD_ + d], sn = sinb[s * HD_ + d];
    const float rv = (d < 64) ? (v * cs - o * sn) : (v * cs + o * sn);
    Kh[((b * G_ + g) * S_ + s) * HD_ + d] = f2bf(rv);
  } else {
    const int cc = c - H_ * HD_ - G_ * HD_;
    const int g = cc >> 7, d = cc & 127;
    Vh[((b * G_ + g) * HD_ + d) * S_ + s] = row[c];   // transposed store
  }
}

// ------------------------- flash attention (causal) -------------------------
// 128 threads = 4 waves; 64-query tile per block, 16 query rows per wave.
// K tiles [key][hd] and V^T tiles [hd][key] stream through double-buffered
// LDS via async b128 copies. Online softmax with wave-half reductions; P goes
// through per-wave LDS to convert C-layout -> A-layout for P*V.
__global__ __launch_bounds__(128)
void flash_attn_kernel(const bf16* __restrict__ Q, const bf16* __restrict__ Kh,
                       const bf16* __restrict__ Vh, bf16* __restrict__ ctx) {
  constexpr int LDK = 136;   // [key][hd] stride, 128 + 8 pad (16B-aligned)
  constexpr int LDV = 72;    // [hd][key] stride,  64 + 8 pad
  constexpr int LDP = 72;
  __shared__ alignas(16) bf16 Ks[2][64 * LDK];
  __shared__ alignas(16) bf16 Vt[2][128 * LDV];
  __shared__ alignas(16) bf16 Ps[4][16 * LDP];

  const int tid  = threadIdx.x;
  const int lane = tid & 31;
  const int half = (lane >> 4) & 1;
  const int lm   = lane & 15;
  const int wv   = tid >> 5;
  const int qt   = blockIdx.x;
  const int bh   = blockIdx.y;
  const int b = bh / H_, h = bh % H_;
  const int g = h / GROUP_;
  const int q0 = qt * 64;

  const bf16* Qbase = Q  + ((b * H_ + h) * S_) * HD_;
  const bf16* Kbase = Kh + ((b * G_ + g) * S_) * HD_;
  const bf16* Vbase = Vh + ((b * G_ + g) * HD_) * S_;   // [hd][S]

  // Q fragments held in registers for the whole pass (A layout, 4 K-steps).
  v16bf qf[4];
  {
    const bf16* qrow = Qbase + (q0 + wv * 16 + lm) * HD_;
    #pragma unroll
    for (int kk = 0; kk < 4; ++kk) qf[kk] = load_frag_a(qrow + kk * 32, half);
  }

  // Each tile is 64*128*2B = 16KB = 1024 x 16B chunks; 8 chunks/thread each.
  auto issue_kv = [&](int kt, int buf) {
    #pragma unroll
    for (int u = 0; u < 8; ++u) {
      const int c = u * 128 + tid;
      const int krow = c >> 4, koff = (c & 15) * 8;
      async_copy16(Kbase + (kt * 64 + krow) * HD_ + koff,
                   &Ks[buf][krow * LDK + koff]);
    }
    #pragma unroll
    for (int u = 0; u < 8; ++u) {
      const int c = u * 128 + tid;
      const int vrow = c >> 3, voff = (c & 7) * 8;
      async_copy16(Vbase + vrow * S_ + kt * 64 + voff,
                   &Vt[buf][vrow * LDV + voff]);
    }
  };

  const v8f vzero = {};
  v8f acc_o[8];
  #pragma unroll
  for (int t = 0; t < 8; ++t) acc_o[t] = vzero;
  float mrow[8], lrow[8];
  #pragma unroll
  for (int r = 0; r < 8; ++r) { mrow[r] = -1e30f; lrow[r] = 0.f; }

  issue_kv(0, 0);
  for (int kt = 0; kt <= qt; ++kt) {
    const int cb = kt & 1;
    wait_async_all();
    __syncthreads();                         // tile kt visible to all waves
    if (kt < qt) issue_kv(kt + 1, cb ^ 1);   // overlap next copy with compute

    // scores S = Q (16x128) * K^T (128x64)
    v8f sacc[4];
    #pragma unroll
    for (int nt = 0; nt < 4; ++nt) sacc[nt] = vzero;
    #pragma unroll
    for (int kk = 0; kk < 4; ++kk) {
      #pragma unroll
      for (int nt = 0; nt < 4; ++nt) {
        v16bf bk = load_frag_b(&Ks[cb][(nt * 16 + lm) * LDK + kk * 32], half);
        sacc[nt] = __builtin_amdgcn_wmma_f32_16x16x32_bf16(
            false, qf[kk], false, bk, (short)0, sacc[nt], false, false);
      }
    }

    if (kt == qt) {                          // causal mask on diagonal tile
      #pragma unroll
      for (int nt = 0; nt < 4; ++nt) {
        const int kidx = kt * 64 + nt * 16 + lm;
        #pragma unroll
        for (int r = 0; r < 8; ++r) {
          const int qidx = q0 + wv * 16 + r + 8 * half;
          sacc[nt][r] = (kidx > qidx) ? -1e30f : sacc[nt][r];
        }
      }
    }

    // online softmax update
    float corr[8];
    #pragma unroll
    for (int r = 0; r < 8; ++r) {
      float v = sacc[0][r];
      v = fmaxf(v, sacc[1][r]); v = fmaxf(v, sacc[2][r]); v = fmaxf(v, sacc[3][r]);
      const float mnew = fmaxf(mrow[r], half_max(v));
      corr[r] = __expf(mrow[r] - mnew);
      mrow[r] = mnew;
      lrow[r] *= corr[r];
    }
    #pragma unroll
    for (int t = 0; t < 8; ++t)
      #pragma unroll
      for (int r = 0; r < 8; ++r) acc_o[t][r] *= corr[r];

    float psum[8];
    #pragma unroll
    for (int r = 0; r < 8; ++r) psum[r] = 0.f;
    #pragma unroll
    for (int nt = 0; nt < 4; ++nt) {
      #pragma unroll
      for (int r = 0; r < 8; ++r) {
        const float p = __expf(sacc[nt][r] - mrow[r]);
        psum[r] += p;
        Ps[wv][(r + 8 * half) * LDP + nt * 16 + lm] = f2bf(p);
      }
    }
    #pragma unroll
    for (int r = 0; r < 8; ++r) lrow[r] += half_sum(psum[r]);

    // O += P (16x64) * V (64x128)
    const v16bf pf0 = load_frag_a(&Ps[wv][lm * LDP + 0],  half);
    const v16bf pf1 = load_frag_a(&Ps[wv][lm * LDP + 32], half);
    #pragma unroll
    for (int ht = 0; ht < 8; ++ht) {
      v16bf bv0 = load_frag_b(&Vt[cb][(ht * 16 + lm) * LDV + 0],  half);
      acc_o[ht] = __builtin_amdgcn_wmma_f32_16x16x32_bf16(
          false, pf0, false, bv0, (short)0, acc_o[ht], false, false);
      v16bf bv1 = load_frag_b(&Vt[cb][(ht * 16 + lm) * LDV + 32], half);
      acc_o[ht] = __builtin_amdgcn_wmma_f32_16x16x32_bf16(
          false, pf1, false, bv1, (short)0, acc_o[ht], false, false);
    }
  }

  // normalize and write ctx[b*S + q][h*128 + hd] (token-major for out-proj)
  #pragma unroll
  for (int ht = 0; ht < 8; ++ht) {
    #pragma unroll
    for (int r = 0; r < 8; ++r) {
      const int qi  = q0 + wv * 16 + r + 8 * half;
      const float val = acc_o[ht][r] / lrow[r];
      ctx[(b * S_ + qi) * (H_ * HD_) + h * HD_ + ht * 16 + lm] = f2bf(val);
    }
  }
}

// -------------------------------- launcher ----------------------------------
extern "C" void kernel_launch(void* const* d_in, const int* in_sizes, int n_in,
                              void* d_out, int out_size, void* d_ws, size_t ws_size,
                              hipStream_t stream) {
  (void)in_sizes; (void)n_in; (void)out_size; (void)ws_size;
  const float* x    = (const float*)d_in[0];
  const float* Wq   = (const float*)d_in[1];
  const float* Wk   = (const float*)d_in[2];
  const float* Wv   = (const float*)d_in[3];
  const float* Wo   = (const float*)d_in[4];
  const float* cosb = (const float*)d_in[5];
  const float* sinb = (const float*)d_in[6];
  // d_in[7] = mask (causal, implemented in-kernel), d_in[8] = start_pos (0)

  char* ws = (char*)d_ws;
  size_t off = 0;
  auto alloc = [&](size_t bytes) -> void* {
    void* p = ws + off; off += (bytes + 255) & ~(size_t)255; return p;
  };
  bf16* xb    = (bf16*)alloc((size_t)B_ * S_ * D_ * 2);       // 33.5 MB
  bf16* wcatT = (bf16*)alloc((size_t)NQKV_ * D_ * 2);         // 50.3 MB  [N][K]
  bf16* wobT  = (bf16*)alloc((size_t)D_ * D_ * 2);            // 33.5 MB  [N][K]
  bf16* qkv   = (bf16*)alloc((size_t)B_ * S_ * NQKV_ * 2);    // 50.3 MB
  bf16* Qb    = (bf16*)alloc((size_t)B_ * H_ * S_ * HD_ * 2); // 33.5 MB
  bf16* Kb    = (bf16*)alloc((size_t)B_ * G_ * S_ * HD_ * 2); //  8.4 MB
  bf16* Vb    = (bf16*)alloc((size_t)B_ * G_ * S_ * HD_ * 2); //  8.4 MB
  bf16* ctx   = xb;  // xb dead after QKV GEMM; reuse for ctx

  {   // x -> bf16 (row-major)
    int n = B_ * S_ * D_;
    hipLaunchKernelGGL(cvt_kernel, dim3((n + 255) / 256), dim3(256), 0, stream,
                       x, xb, n);
  }
  auto cvtT = [&](const float* src, bf16* dst, int K, int N, int ldk, int roff) {
    long long n = (long long)K * N;
    hipLaunchKernelGGL(cvtT_kernel, dim3((unsigned)((n + 255) / 256)), dim3(256),
                       0, stream, src, dst, K, N, ldk, roff);
  };
  cvtT(Wq, wcatT, D_, H_ * HD_, D_, 0);
  cvtT(Wk, wcatT, D_, G_ * HD_, D_, H_ * HD_);
  cvtT(Wv, wcatT, D_, G_ * HD_, D_, H_ * HD_ + G_ * HD_);
  cvtT(Wo, wobT,  H_ * HD_, D_, H_ * HD_, 0);

  // QKV = x @ [Wq|Wk|Wv]  (M=4096, N=6144, K=4096), bf16 out
  hipLaunchKernelGGL(gemm_bf16_kernel, dim3(NQKV_ / 128, (B_ * S_) / 128), dim3(256),
                     0, stream, xb, wcatT, (void*)qkv, B_ * S_, NQKV_, D_, 0);

  // RoPE + scale + head-major scatter (V stored hd-major)
  {
    long long n = (long long)B_ * S_ * NQKV_;
    hipLaunchKernelGGL(rope_scatter_kernel, dim3((unsigned)((n + 255) / 256)),
                       dim3(256), 0, stream, qkv, cosb, sinb, Qb, Kb, Vb);
  }

  // causal flash attention -> ctx (bf16, token-major)
  hipLaunchKernelGGL(flash_attn_kernel, dim3(S_ / 64, B_ * H_), dim3(128),
                     0, stream, Qb, Kb, Vb, ctx);

  // out = ctx @ Wo  (M=4096, N=4096, K=4096), fp32 out
  hipLaunchKernelGGL(gemm_bf16_kernel, dim3(D_ / 128, (B_ * S_) / 128), dim3(256),
                     0, stream, ctx, wobT, d_out, B_ * S_, D_, H_ * HD_, 1);
}
